// GeometricAttention_79370995631008
// MI455X (gfx1250) — matrix operand order repaired and checked by reading, not compiled, not run on hardware
//
#include <hip/hip_runtime.h>

// ---------------------------------------------------------------- constants
constexpr int Bsz = 2, S = 2048, D = 1024, H = 4, DK = 256;
constexpr int BHS = Bsz * H * S;

typedef __bf16 bf16_t;
typedef bf16_t v16bf __attribute__((ext_vector_type(16)));
typedef bf16_t v8bf  __attribute__((ext_vector_type(8)));
typedef float  v8f   __attribute__((ext_vector_type(8)));

__device__ __forceinline__ v8f wmma_bf16(v16bf a, v16bf b, v8f c) {
    return __builtin_amdgcn_wmma_f32_16x16x32_bf16(
        false, a, false, b, (short)0, c, false, false);
}

__device__ __forceinline__ void async_b128(unsigned lds_off, unsigned long long gaddr) {
    asm volatile("global_load_async_to_lds_b128 %0, %1, off"
                 :: "v"(lds_off), "v"(gaddr) : "memory");
}

__device__ __forceinline__ v16bf cvt16(const float* p, int o1) {
    // elements 0..7 from p[0..8), 8..15 from p[o1..o1+8)
    float4 a0 = *(const float4*)(p);
    float4 a1 = *(const float4*)(p + 4);
    float4 a2 = *(const float4*)(p + o1);
    float4 a3 = *(const float4*)(p + o1 + 4);
    v16bf v;
    v[0]=(bf16_t)a0.x; v[1]=(bf16_t)a0.y; v[2]=(bf16_t)a0.z; v[3]=(bf16_t)a0.w;
    v[4]=(bf16_t)a1.x; v[5]=(bf16_t)a1.y; v[6]=(bf16_t)a1.z; v[7]=(bf16_t)a1.w;
    v[8]=(bf16_t)a2.x; v[9]=(bf16_t)a2.y; v[10]=(bf16_t)a2.z; v[11]=(bf16_t)a2.w;
    v[12]=(bf16_t)a3.x; v[13]=(bf16_t)a3.y; v[14]=(bf16_t)a3.z; v[15]=(bf16_t)a3.w;
    return v;
}

// ---------------------------------------------------------------- GEMM
// C[M,N] = A[M,K] @ Bw[N,K]^T + bias[N]  (f32 io, bf16 WMMA compute)
// 128 threads = 4 waves (2x2); 64x64 block tile; K-step 32.
// Double-buffered LDS fed by global_load_async_to_lds_b128 (ASYNCcnt).
constexpr int GP = 36;  // LDS row pitch in floats (144B: 16B aligned, bank-spread)

__global__ __launch_bounds__(128, 1) void gemm_bias_kernel(
    const float* __restrict__ A, const float* __restrict__ Bw,
    const float* __restrict__ bias, float* __restrict__ C,
    int M, int N, int K)
{
    __shared__ float ldsA[2][64 * GP];
    __shared__ float ldsB[2][64 * GP];

    const int tid  = threadIdx.x;
    const int wave = tid >> 5;
    const int lane = tid & 31;
    const int l16  = lane & 15;
    const int half = lane >> 4;
    const int wm = wave >> 1, wn = wave & 1;

    const int nbn = N / 64;
    const int mbase = (blockIdx.x / nbn) * 64;
    const int nbase = (blockIdx.x % nbn) * 64;
    const int nk = K / 32;

    const unsigned la0 = (unsigned)(unsigned long long)&ldsA[0][0];
    const unsigned lb0 = (unsigned)(unsigned long long)&ldsB[0][0];
    const unsigned lbufstride = (unsigned)(64 * GP * 4);

    auto stage = [&](int kt, int buf) {
        const unsigned long long krow = (unsigned long long)kt * 32ull;
#pragma unroll
        for (int i = 0; i < 4; ++i) {               // A tile: 64 rows x 128B
            int chunk = i * 128 + tid;              // 0..511
            int r  = chunk >> 3;
            int cb = (chunk & 7) * 16;
            async_b128(la0 + buf * lbufstride + (unsigned)(r * GP * 4 + cb),
                       (unsigned long long)(A + (size_t)(mbase + r) * K + krow) + cb);
        }
#pragma unroll
        for (int i = 0; i < 4; ++i) {               // B tile: 64 rows x 128B
            int chunk = i * 128 + tid;
            int r  = chunk >> 3;
            int cb = (chunk & 7) * 16;
            async_b128(lb0 + buf * lbufstride + (unsigned)(r * GP * 4 + cb),
                       (unsigned long long)(Bw + (size_t)(nbase + r) * K + krow) + cb);
        }
    };

    v8f acc[2][2] = {};
    stage(0, 0);

    for (int kt = 0; kt < nk; ++kt) {
        const int buf = kt & 1;
        if (kt + 1 < nk) {
            stage(kt + 1, buf ^ 1);
            asm volatile("s_wait_asynccnt 8" ::: "memory");   // retire current stage
        } else {
            asm volatile("s_wait_asynccnt 0" ::: "memory");
        }
        __syncthreads();

        v16bf av[2], bv[2];
#pragma unroll
        for (int ms = 0; ms < 2; ++ms)   // A: 16x32. lane<16: K0..7 & 16..23 ; lane>=16: +8
            av[ms] = cvt16(&ldsA[buf][(wm * 32 + ms * 16 + l16) * GP + half * 8], 16);
#pragma unroll
        for (int ns = 0; ns < 2; ++ns) { // B: 32x16. lane<16: K0..7 & 16..23 ; lane>=16: +8
            const float* bp = &ldsB[buf][(wn * 32 + ns * 16 + l16) * GP + half * 8];
            bv[ns] = cvt16(bp, 16);
        }
#pragma unroll
        for (int ms = 0; ms < 2; ++ms)
#pragma unroll
            for (int ns = 0; ns < 2; ++ns)
                acc[ms][ns] = wmma_bf16(av[ms], bv[ns], acc[ms][ns]);
        __syncthreads();
    }

#pragma unroll
    for (int ms = 0; ms < 2; ++ms)
#pragma unroll
        for (int ns = 0; ns < 2; ++ns) {
            const int n = nbase + wn * 32 + ns * 16 + l16;
            const float bia = bias[n];
#pragma unroll
            for (int r = 0; r < 8; ++r) {
                const int m = mbase + wm * 32 + ms * 16 + half * 8 + r;
                C[(size_t)m * N + n] = acc[ms][ns][r] + bia;
            }
        }
}

// ---------------------------------------------------------------- psi
__global__ __launch_bounds__(256) void psi_kernel(
    const float* __restrict__ X, const float* __restrict__ projw,
    const float* __restrict__ projb,
    bf16_t* __restrict__ out, bf16_t* __restrict__ outsp,
    float* __restrict__ n2out)
{
    const int wv = blockIdx.x * 8 + (threadIdx.x >> 5);
    if (wv >= BHS) return;
    const int lane = threadIdx.x & 31;
    const int b = wv / (H * S);
    const int h = (wv / S) % H;
    const int s = wv % S;

    const float* x = X + ((size_t)(b * S + s)) * D + h * DK + lane * 8;
    float xv[8];
    float4 x0 = *(const float4*)(x);
    float4 x1 = *(const float4*)(x + 4);
    xv[0]=x0.x; xv[1]=x0.y; xv[2]=x0.z; xv[3]=x0.w;
    xv[4]=x1.x; xv[5]=x1.y; xv[6]=x1.z; xv[7]=x1.w;

    float gen[24];
#pragma unroll
    for (int j = 0; j < 24; ++j) {
        const float* w = projw + j * DK + lane * 8;
        float4 w0 = *(const float4*)(w);
        float4 w1 = *(const float4*)(w + 4);
        float sum = xv[0]*w0.x + xv[1]*w0.y + xv[2]*w0.z + xv[3]*w0.w
                  + xv[4]*w1.x + xv[5]*w1.y + xv[6]*w1.z + xv[7]*w1.w;
        sum += __shfl_xor(sum, 16, 32);
        sum += __shfl_xor(sum, 8, 32);
        sum += __shfl_xor(sum, 4, 32);
        sum += __shfl_xor(sum, 2, 32);
        sum += __shfl_xor(sum, 1, 32);
        gen[j] = sum + projb[j];
    }

    float pr[8], pim[8], n2 = 0.f;
#pragma unroll
    for (int i = 0; i < 8; ++i) {
        float th = fminf(fmaxf(gen[16 + i], -8.0f), 8.0f);
        float e = __expf(th), c = __cosf(th), sn = __sinf(th);
        pr[i]  = gen[i] * e + gen[8 + i] * c;
        pim[i] = gen[8 + i] * sn;
        n2 += pr[i] * pr[i] + pim[i] * pim[i];
    }

    if (lane == 0) {
        bf16_t* o = out + (size_t)wv * 16;
#pragma unroll
        for (int i = 0; i < 8; ++i) { o[i] = (bf16_t)pr[i]; o[8 + i] = (bf16_t)pim[i]; }
        n2out[wv] = n2;
        if (outsp) {
            bf16_t* osp = outsp + (size_t)wv * 16;
#pragma unroll
            for (int i = 0; i < 8; ++i) { osp[i] = (bf16_t)pim[i]; osp[8 + i] = (bf16_t)(-pr[i]); }
        }
    }
}

// ---------------------------------------------------------------- fused attention
// grid = (S/128, B*H, 2). Wave w of 8 owns 16 query rows; blockIdx.z picks
// which 128 of the 256 head-dims this WG accumulates (halves VGPR pressure).
__global__ __launch_bounds__(256, 1) void geo_attn_kernel(
    const bf16_t* __restrict__ pq, const bf16_t* __restrict__ pk,
    const bf16_t* __restrict__ pksp,
    const float* __restrict__ qn2, const float* __restrict__ kn2,
    const float* __restrict__ V, float* __restrict__ AT,
    const float* __restrict__ geo, const float* __restrict__ temp)
{
    constexpr int VP = 40;                 // padded V^T pitch (rows)
    __shared__ bf16_t lds_vt[128][VP];     // V^T tile: [col][row]
    __shared__ bf16_t lds_p[8][16][32];    // per-wave P tile

    const int tid  = threadIdx.x;
    const int wave = tid >> 5;
    const int lane = tid & 31;
    const int l16  = lane & 15;
    const int half = lane >> 4;
    const int bh   = blockIdx.y;
    const int b    = bh / H;
    const int h    = bh % H;
    const int dh   = blockIdx.z;           // 0/1: which 128 dims
    const int qbase = blockIdx.x * 128 + wave * 16;

    const float g0 = geo[0], g1 = geo[1], g2 = geo[2], g3 = geo[3];
    const float invT = 0.25f / temp[0];

    v16bf a_pq = {};
    {
        const bf16_t* p = pq + ((size_t)bh * S + qbase + l16) * 16;
        if (half == 0) a_pq = *(const v16bf*)p;
    }

    float qn[8], sqq[8];
#pragma unroll
    for (int r = 0; r < 8; ++r) {
        qn[r]  = qn2[(size_t)bh * S + qbase + half * 8 + r];
        sqq[r] = sqrtf(qn[r] + 1e-6f);
    }

    float mrun[8], lrun[8];
#pragma unroll
    for (int r = 0; r < 8; ++r) { mrun[r] = -1e30f; lrun[r] = 0.f; }
    v8f acc[8] = {};

    for (int kb = 0; kb < S; kb += 32) {
        __syncthreads();
        {   // stage V^T: 32 rows x 128 cols (this WG's half of DK)
            const int c  = tid & 127;
            const int rb = (tid >> 7) * 16;
            const float* vsrc = V + ((size_t)(b * S + kb + rb)) * D + h * DK + dh * 128 + c;
#pragma unroll
            for (int i = 0; i < 16; ++i)
                lds_vt[c][rb + i] = (bf16_t)vsrc[(size_t)i * D];
        }
        __syncthreads();

        v16bf bpk0 = {}, bpk1 = {}, bsp0 = {}, bsp1 = {};
        if (half == 0) {
            const size_t r0 = ((size_t)bh * S + kb + l16) * 16;
            bpk0 = *(const v16bf*)(pk + r0);
            bsp0 = *(const v16bf*)(pksp + r0);
            bpk1 = *(const v16bf*)(pk + r0 + 256);
            bsp1 = *(const v16bf*)(pksp + r0 + 256);
        }
        v8f z = {};
        v8f tn0 = wmma_bf16(a_pq, bpk0, z);
        v8f sp0 = wmma_bf16(a_pq, bsp0, z);
        v8f tn1 = wmma_bf16(a_pq, bpk1, z);
        v8f sp1 = wmma_bf16(a_pq, bsp1, z);

        const float kn0 = kn2[(size_t)bh * S + kb + l16];
        const float kn1 = kn2[(size_t)bh * S + kb + 16 + l16];
        const float sqk0 = sqrtf(kn0 + 1e-6f);
        const float sqk1 = sqrtf(kn1 + 1e-6f);

        float alpha[8];
#pragma unroll
        for (int r = 0; r < 8; ++r) {
            float t0 = tn0[r], s0 = sp0[r];
            float t1 = tn1[r], s1 = sp1[r];
            float w0 = sqrtf(fmaxf(qn[r] * kn0 - t0 * t0 - s0 * s0, 0.f) + 1e-6f);
            float w1 = sqrtf(fmaxf(qn[r] * kn1 - t1 * t1 - s1 * s1, 0.f) + 1e-6f);
            float sc0 = (g0 * w0 + g1 * t0 + g2 * s0 + g3 * sqq[r] * sqk0) * invT;
            float sc1 = (g0 * w1 + g1 * t1 + g2 * s1 + g3 * sqq[r] * sqk1) * invT;

            float mx = fmaxf(sc0, sc1);
            mx = fmaxf(mx, __shfl_xor(mx, 1, 32));
            mx = fmaxf(mx, __shfl_xor(mx, 2, 32));
            mx = fmaxf(mx, __shfl_xor(mx, 4, 32));
            mx = fmaxf(mx, __shfl_xor(mx, 8, 32));
            const float mnew = fmaxf(mrun[r], mx);
            const float al = __expf(mrun[r] - mnew);
            const float p0 = __expf(sc0 - mnew);
            const float p1 = __expf(sc1 - mnew);
            float ps = p0 + p1;
            ps += __shfl_xor(ps, 1, 32);
            ps += __shfl_xor(ps, 2, 32);
            ps += __shfl_xor(ps, 4, 32);
            ps += __shfl_xor(ps, 8, 32);
            lrun[r] = lrun[r] * al + ps;
            mrun[r] = mnew;
            alpha[r] = al;
            lds_p[wave][half * 8 + r][l16]      = (bf16_t)p0;
            lds_p[wave][half * 8 + r][16 + l16] = (bf16_t)p1;
        }

#pragma unroll
        for (int nt = 0; nt < 8; ++nt)
#pragma unroll
            for (int r = 0; r < 8; ++r) acc[nt][r] *= alpha[r];

        v16bf ap;
        {
            v8bf lo = *(const v8bf*)&lds_p[wave][l16][half * 16];
            v8bf hi = *(const v8bf*)&lds_p[wave][l16][half * 16 + 8];
#pragma unroll
            for (int i = 0; i < 8; ++i) { ap[i] = lo[i]; ap[8 + i] = hi[i]; }
        }

#pragma unroll
        for (int nt = 0; nt < 8; ++nt) {
            v16bf bv = *(const v16bf*)&lds_vt[nt * 16 + l16][half * 16];
            acc[nt] = wmma_bf16(ap, bv, acc[nt]);
        }
    }

    float invl[8];
#pragma unroll
    for (int r = 0; r < 8; ++r) invl[r] = 1.0f / lrun[r];
#pragma unroll
    for (int nt = 0; nt < 8; ++nt)
#pragma unroll
        for (int r = 0; r < 8; ++r) {
            const int m = qbase + half * 8 + r;
            AT[((size_t)(b * S + m)) * D + h * DK + dh * 128 + nt * 16 + l16] =
                acc[nt][r] * invl[r];
        }
}

// ---------------------------------------------------------------- launch
extern "C" void kernel_launch(void* const* d_in, const int* in_sizes, int n_in,
                              void* d_out, int out_size, void* d_ws, size_t ws_size,
                              hipStream_t stream) {
    const float* Qin  = (const float*)d_in[0];
    const float* Kin  = (const float*)d_in[1];
    const float* Vin  = (const float*)d_in[2];
    // d_in[3] = T_field (unused by reference)
    const float* Wq_w = (const float*)d_in[4];
    const float* Wq_b = (const float*)d_in[5];
    const float* Wo_w = (const float*)d_in[6];
    const float* Wo_b = (const float*)d_in[7];
    const float* proj_w = (const float*)d_in[8];
    const float* proj_b = (const float*)d_in[9];
    const float* geo    = (const float*)d_in[10];
    const float* temp   = (const float*)d_in[11];
    float* out = (float*)d_out;

    char* ws = (char*)d_ws;
    float*  Qp   = (float*)ws;                       // B*S*D f32
    float*  AT   = Qp + (size_t)Bsz * S * D;         // B*S*D f32
    bf16_t* pq   = (bf16_t*)(AT + (size_t)Bsz * S * D);
    bf16_t* pk   = pq + (size_t)BHS * 16;
    bf16_t* pksp = pk + (size_t)BHS * 16;
    float*  qn2  = (float*)(pksp + (size_t)BHS * 16);
    float*  kn2  = qn2 + BHS;

    const int M = Bsz * S, N = D, K = D;
    const int gemm_wgs = (M / 64) * (N / 64);        // 1024
    const int psi_wgs  = BHS / 8;                    // 2048

    gemm_bias_kernel<<<gemm_wgs, 128, 0, stream>>>(Qin, Wq_w, Wq_b, Qp, M, N, K);
    psi_kernel<<<psi_wgs, 256, 0, stream>>>(Qp, proj_w, proj_b, pq, nullptr, qn2);
    psi_kernel<<<psi_wgs, 256, 0, stream>>>(Kin, proj_w, proj_b, pk, pksp, kn2);
    geo_attn_kernel<<<dim3(S / 128, Bsz * H, 2), 256, 0, stream>>>(
        pq, pk, pksp, qn2, kn2, Vin, AT, geo, temp);
    gemm_bias_kernel<<<gemm_wgs, 128, 0, stream>>>(AT, Wo_w, Wo_b, out, M, N, K);
}
